// CrossAttention_60722247631238
// MI455X (gfx1250) — matrix operand order repaired and checked by reading, not compiled, not run on hardware
//
#include <hip/hip_runtime.h>
#include <hip/hip_bf16.h>

// ---------------------------------------------------------------------------
// Types
// ---------------------------------------------------------------------------
typedef __bf16 bf16;
typedef bf16  v16bf __attribute__((ext_vector_type(16)));
typedef bf16  v8bf  __attribute__((ext_vector_type(8)));
typedef float v8f   __attribute__((ext_vector_type(8)));
typedef unsigned int u32x4 __attribute__((ext_vector_type(4)));
typedef int i32x8 __attribute__((ext_vector_type(8)));
typedef int i32x4 __attribute__((ext_vector_type(4)));

#define SCALE_ATTN 0.125f  // 64^-0.5

// ---------------------------------------------------------------------------
// Fragment helpers (CDNA5 WMMA bf16 16x16x32 layouts, wave32)
//
// A (16x32, MxK): lane L: row = L&15; half h = L>>4;
//   elems 0..7  = A[row][h*8 + 0..7]
//   elems 8..15 = A[row][h*8 + 16..23]
// B (32x16, KxN) loaded from a B^T-layout buffer (rows = N, stride over K):
//   lane L: col = L&15; half h = L>>4;
//   elems 0..7 = BT[col][h*8 + 0..7], elems 8..15 = BT[col][h*8 + 16..23]
// ---------------------------------------------------------------------------
__device__ __forceinline__ v16bf frag_cat(v8bf lo, v8bf hi) {
  return __builtin_shufflevector(lo, hi, 0, 1, 2, 3, 4, 5, 6, 7,
                                 8, 9, 10, 11, 12, 13, 14, 15);
}

__device__ __forceinline__ v16bf load_a_frag(const bf16* base, int ld,
                                             int row0, int k0) {
  const int lane = threadIdx.x & 31;
  const int r = lane & 15, h = lane >> 4;
  const bf16* p = base + (size_t)(row0 + r) * ld + k0 + h * 8;
  v8bf lo = *(const v8bf*)(p);
  v8bf hi = *(const v8bf*)(p + 16);
  return frag_cat(lo, hi);
}

__device__ __forceinline__ v16bf load_bt_frag(const bf16* base, int ld,
                                              int col0, int k0) {
  const int lane = threadIdx.x & 31;
  const int r = lane & 15, h = lane >> 4;
  const bf16* p = base + (size_t)(col0 + r) * ld + k0 + h * 8;
  v8bf lo = *(const v8bf*)(p);
  v8bf hi = *(const v8bf*)(p + 16);
  return frag_cat(lo, hi);
}

__device__ __forceinline__ v8f wmma_bf16(v16bf a, v16bf b, v8f c) {
  // (neg_a, A, neg_b, B, c_mod, C, reuse_a, reuse_b)
  return __builtin_amdgcn_wmma_f32_16x16x32_bf16(false, a, false, b,
                                                 (short)0, c, false, false);
}

__device__ __forceinline__ float rmax16(float v) {
#pragma unroll
  for (int m = 8; m >= 1; m >>= 1) v = fmaxf(v, __shfl_xor(v, m, 32));
  return v;
}
__device__ __forceinline__ float rsum16(float v) {
#pragma unroll
  for (int m = 8; m >= 1; m >>= 1) v += __shfl_xor(v, m, 32);
  return v;
}

// ---------------------------------------------------------------------------
// Tensor Data Mover: 2D tile load global -> LDS (D# per CDNA5 ISA).
// All args wave-uniform. tile/tensor dims & stride in elements (2B each).
// group0: [1:0]=count=1 | lds_addr | global_addr[56:0] | type=2
// group1: data_size=1(2B) | tensor_dim0/1 | tile_dim0/1 | dim0_stride
// groups 2/3 + trailing descriptor words zero (<=2D tensor).
// This toolchain exposes the 6-arg builtin:
//   (u32x4 g0, i32x8 g1, i32x4 g2, i32x4 g3, i32x8 g4, i32 cpol)
// ---------------------------------------------------------------------------
__device__ __forceinline__ void tdm_load_2d(unsigned lds_off, const void* gptr,
                                            unsigned tile_w, unsigned tile_h,
                                            unsigned stride_elems,
                                            unsigned tensor_w,
                                            unsigned tensor_h) {
  unsigned long long ga = (unsigned long long)(uintptr_t)gptr;
  u32x4 g0;
  g0[0] = 1u;                                         // count=1, user mode
  g0[1] = lds_off;                                    // lds_addr (bytes)
  g0[2] = (unsigned)(ga & 0xffffffffu);               // global_addr[31:0]
  g0[3] = (unsigned)((ga >> 32) & 0x01ffffffu) | (2u << 30);  // [56:32]|type=2
  i32x8 g1;
  g1[0] = 0x00010000;                                 // data_size=1 (2 bytes)
  g1[1] = (int)((tensor_w & 0xffffu) << 16);          // tensor_dim0[15:0]
  g1[2] = (int)(((tensor_w >> 16) & 0xffffu) | ((tensor_h & 0xffffu) << 16));
  g1[3] = (int)(((tensor_h >> 16) & 0xffffu) | ((tile_w & 0xffffu) << 16));
  g1[4] = (int)(tile_h & 0xffffu);                    // tile_dim1, tile_dim2=0
  g1[5] = (int)stride_elems;                          // dim0_stride[31:0]
  g1[6] = 0;                                          // stride hi, dim1_stride
  g1[7] = 0;
  i32x4 g2 = {0, 0, 0, 0};
  i32x4 g3 = {0, 0, 0, 0};
  i32x8 g4 = {0, 0, 0, 0, 0, 0, 0, 0};
  __builtin_amdgcn_tensor_load_to_lds(g0, g1, g2, g3, g4, 0);
}

// ---------------------------------------------------------------------------
// fp32 -> bf16 conversion (plain)
// ---------------------------------------------------------------------------
__global__ void f32_to_bf16_kernel(const float* __restrict__ in,
                                   bf16* __restrict__ out, int n) {
  int i = blockIdx.x * blockDim.x + threadIdx.x;
  int stride = gridDim.x * blockDim.x;
  for (; i < n; i += stride) out[i] = (bf16)in[i];
}

// fp32 (K,N) -> bf16 transposed (N,K). One-time cost per weight matrix so the
// GEMM hot loop needs no LDS transposes and both tiles are TDM-friendly.
__global__ void f32_to_bf16_t_kernel(const float* __restrict__ in,
                                     bf16* __restrict__ out, int K, int N) {
  int idx = blockIdx.x * blockDim.x + threadIdx.x;
  int total = K * N;
  int stride = gridDim.x * blockDim.x;
  for (; idx < total; idx += stride) {
    int n = idx / K, k = idx - n * K;           // output (n, k), coalesced-ish
    out[idx] = (bf16)in[(size_t)k * N + n];
  }
}

// ---------------------------------------------------------------------------
// bf16 GEMM with TDM double buffering: C[M,N] = A[M,K] @ BT[N,K]^T + bias[N]
// Block: 128x64 tile, 256 threads (8 waves, wave = 32x32 = 2x2 WMMA tiles),
// K-step 64 (2 WMMA k-frags). Wave 0 drives the Tensor Data Mover; TENSORcnt
// completes in-order so waiting for <=2 releases the current buffer while the
// next pair streams in.
// M % 128 == 0, N % 64 == 0, K % 64 == 0.
// ---------------------------------------------------------------------------
template <typename OutT>
__global__ __launch_bounds__(256) void gemm_bf16_tdm_kernel(
    const bf16* __restrict__ A, const bf16* __restrict__ BT,
    const float* __restrict__ bias, OutT* __restrict__ C, int M, int N,
    int K) {
  __shared__ bf16 sA[2][128 * 64];    // 2 x 16 KB
  __shared__ bf16 sBt[2][64 * 64];    // 2 x  8 KB

  const int tid = threadIdx.x;
  const int lane = tid & 31;
  const int wid = tid >> 5;
  const int wm = wid >> 1;  // 0..3
  const int wn = wid & 1;   // 0..1
  const int row0 = blockIdx.y * 128;
  const int col0 = blockIdx.x * 64;
  const int nk = K >> 6;

  const bf16* Atile = A + (size_t)row0 * K;
  const bf16* Btile = BT + (size_t)col0 * K;

  if (tid < 32) {  // wave 0: prologue DMA for buffer 0
    tdm_load_2d((unsigned)(uintptr_t)&sA[0][0], Atile, 64, 128,
                (unsigned)K, (unsigned)K, (unsigned)M);
    tdm_load_2d((unsigned)(uintptr_t)&sBt[0][0], Btile, 64, 64,
                (unsigned)K, (unsigned)K, (unsigned)N);
  }

  v8f acc[2][2] = {};

  for (int ik = 0; ik < nk; ++ik) {
    const int cur = ik & 1;
    if (tid < 32) {
      if (ik + 1 < nk) {
        const int k1 = (ik + 1) * 64;
        tdm_load_2d((unsigned)(uintptr_t)&sA[cur ^ 1][0], Atile + k1, 64, 128,
                    (unsigned)K, (unsigned)K, (unsigned)M);
        tdm_load_2d((unsigned)(uintptr_t)&sBt[cur ^ 1][0], Btile + k1, 64, 64,
                    (unsigned)K, (unsigned)K, (unsigned)N);
        __builtin_amdgcn_s_wait_tensorcnt(2);  // current pair done (in-order)
      } else {
        __builtin_amdgcn_s_wait_tensorcnt(0);
      }
    }
    __syncthreads();

#pragma unroll
    for (int kq = 0; kq < 2; ++kq) {
      v16bf af[2], bfv[2];
#pragma unroll
      for (int mi = 0; mi < 2; ++mi)
        af[mi] = load_a_frag(&sA[cur][0], 64, wm * 32 + mi * 16, kq * 32);
#pragma unroll
      for (int ni = 0; ni < 2; ++ni)
        bfv[ni] = load_bt_frag(&sBt[cur][0], 64, wn * 32 + ni * 16, kq * 32);
#pragma unroll
      for (int mi = 0; mi < 2; ++mi)
#pragma unroll
        for (int ni = 0; ni < 2; ++ni)
          acc[mi][ni] = wmma_bf16(af[mi], bfv[ni], acc[mi][ni]);
    }
    __syncthreads();
  }

  // Epilogue: C/D layout -> global. VGPR r: lanes0-15 row=r, lanes16-31 row=8+r.
  const int h = lane >> 4, ln = lane & 15;
#pragma unroll
  for (int mi = 0; mi < 2; ++mi) {
#pragma unroll
    for (int ni = 0; ni < 2; ++ni) {
      const int col = col0 + wn * 32 + ni * 16 + ln;
      const float bv = bias[col];
      const int rb = row0 + wm * 32 + mi * 16 + h * 8;
#pragma unroll
      for (int r = 0; r < 8; ++r) {
        float val = acc[mi][ni][r] + bv;
        C[(size_t)(rb + r) * N + col] = (OutT)val;
      }
    }
  }
}

// ---------------------------------------------------------------------------
// Flash attention over bf16 projections.
//   qp : (B, Nq, C)  bf16, head h occupies cols [h*64, h*64+64)
//   kvp: (B, Nkv, 2C) bf16, K at cols h*64.., V at cols 1024 + h*64..
//   out: (B, Nq, C)  bf16  (== (B, Nq, H, D) flattened, ready for out-proj)
// Block: 128 threads = 4 waves, each wave owns 16 query rows; grid covers
// (Nq/64, H, B). kv processed in blocks of 64 with online softmax.
// ---------------------------------------------------------------------------
__global__ __launch_bounds__(128) void flash_attn_kernel(
    const bf16* __restrict__ qp, const bf16* __restrict__ kvp,
    bf16* __restrict__ attn) {
  constexpr int Nq = 2048, Nkv = 2048, C = 1024, D = 64;

  __shared__ bf16 sVt[64 * 64];      // V^T tile: [d=64][kv=64]
  __shared__ bf16 sP[4][16 * 64];    // per-wave P tile (16 rows x 64 kv)

  const int tid = threadIdx.x;
  const int lane = tid & 31;
  const int w = tid >> 5;
  const int h16 = lane >> 4, ln = lane & 15;
  const int b = blockIdx.z;
  const int hh = blockIdx.y;
  const int q0 = blockIdx.x * 64 + w * 16;

  const bf16* qbase = qp + (size_t)b * Nq * C + hh * D;            // stride C
  const bf16* kbase = kvp + (size_t)b * Nkv * (2 * C) + hh * D;    // stride 2C
  const bf16* vbase = kbase + C;

  // Q fragments (K = d, two k-steps of 32), resident for whole kernel.
  v16bf qf0 = load_a_frag(qbase, C, q0, 0);
  v16bf qf1 = load_a_frag(qbase, C, q0, 32);

  v8f o[4] = {};
  float mrow[8], lrow[8];
#pragma unroll
  for (int r = 0; r < 8; ++r) {
    mrow[r] = -1e30f;
    lrow[r] = 0.0f;
  }

  for (int kv0 = 0; kv0 < Nkv; kv0 += 64) {
    // ---- cooperative stage of V^T tile -------------------------------
#pragma unroll
    for (int i = 0; i < 4; ++i) {
      int c = tid + i * 128;       // 0..511
      int kr = c >> 3;             // kv row in tile, 0..63
      int dg = (c & 7) * 8;        // d group
      v8bf v = *(const v8bf*)(vbase + (size_t)(kv0 + kr) * (2 * C) + dg);
#pragma unroll
      for (int j = 0; j < 8; ++j) sVt[(dg + j) * 64 + kr] = v[j];
    }
    __syncthreads();

    // ---- S = (Q @ K^T) * scale, 4 column tiles of 16 ------------------
    v8f s[4];
#pragma unroll
    for (int nt = 0; nt < 4; ++nt) {
      v16bf kf0 = load_bt_frag(kbase, 2 * C, kv0 + nt * 16, 0);
      v16bf kf1 = load_bt_frag(kbase, 2 * C, kv0 + nt * 16, 32);
      v8f t = {};
      t = wmma_bf16(qf0, kf0, t);
      t = wmma_bf16(qf1, kf1, t);
      s[nt] = t * SCALE_ATTN;
    }

    // ---- online softmax (row stats live per VGPR index r) -------------
#pragma unroll
    for (int r = 0; r < 8; ++r) {
      float mx = fmaxf(fmaxf(s[0][r], s[1][r]), fmaxf(s[2][r], s[3][r]));
      mx = rmax16(mx);
      float mnew = fmaxf(mrow[r], mx);
      float alpha = __expf(mrow[r] - mnew);
      mrow[r] = mnew;
      float rs = 0.0f;
#pragma unroll
      for (int nt = 0; nt < 4; ++nt) {
        float e = __expf(s[nt][r] - mnew);
        s[nt][r] = e;
        rs += e;
      }
      rs = rsum16(rs);
      lrow[r] = lrow[r] * alpha + rs;
#pragma unroll
      for (int dt = 0; dt < 4; ++dt) o[dt][r] *= alpha;
    }

    // ---- P (C-layout) -> per-wave LDS -> A-layout fragments -----------
    bf16* myP = &sP[w][0];
#pragma unroll
    for (int nt = 0; nt < 4; ++nt)
#pragma unroll
      for (int r = 0; r < 8; ++r)
        myP[(h16 * 8 + r) * 64 + nt * 16 + ln] = (bf16)s[nt][r];
    asm volatile("s_wait_dscnt 0" ::: "memory");

    // ---- O += P @ V ----------------------------------------------------
#pragma unroll
    for (int kq = 0; kq < 2; ++kq) {
      v16bf pf = load_a_frag(myP, 64, 0, kq * 32);
#pragma unroll
      for (int dt = 0; dt < 4; ++dt) {
        v16bf vf = load_bt_frag(sVt, 64, dt * 16, kq * 32);
        o[dt] = wmma_bf16(pf, vf, o[dt]);
      }
    }
    __syncthreads();  // sVt consumed before next iteration's stores
  }

  // ---- finalize + store ------------------------------------------------
#pragma unroll
  for (int r = 0; r < 8; ++r) {
    const float inv = 1.0f / lrow[r];
    const int row = q0 + h16 * 8 + r;
#pragma unroll
    for (int dt = 0; dt < 4; ++dt) {
      const int col = hh * D + dt * 16 + ln;
      attn[((size_t)b * Nq + row) * C + col] = (bf16)(o[dt][r] * inv);
    }
  }
}

// ---------------------------------------------------------------------------
// Host launcher
// ---------------------------------------------------------------------------
extern "C" void kernel_launch(void* const* d_in, const int* in_sizes, int n_in,
                              void* d_out, int out_size, void* d_ws,
                              size_t ws_size, hipStream_t stream) {
  (void)in_sizes; (void)n_in; (void)out_size; (void)ws_size;

  const float* q     = (const float*)d_in[0];   // (2, 2048, 1024)
  const float* kv    = (const float*)d_in[1];   // (2, 2048, 1024)
  const float* w_q   = (const float*)d_in[2];   // (1024, 1024)
  const float* b_q   = (const float*)d_in[3];   // (1024,)
  const float* w_kv  = (const float*)d_in[4];   // (1024, 2048)
  const float* b_kv  = (const float*)d_in[5];   // (2048,)
  const float* w_out = (const float*)d_in[6];   // (1024, 1024)
  const float* b_out = (const float*)d_in[7];   // (1024,)
  float* out = (float*)d_out;                   // (2, 2048, 1024)

  const size_t nQ    = (size_t)2 * 2048 * 1024;   // 4M
  const size_t nKV   = nQ;
  const size_t nWq   = (size_t)1024 * 1024;
  const size_t nWkv  = (size_t)1024 * 2048;
  const size_t nWout = (size_t)1024 * 1024;
  const size_t nQP   = nQ;
  const size_t nKVP  = (size_t)2 * 2048 * 2048;
  const size_t nATT  = nQ;

  char* ws = (char*)d_ws;
  bf16* q_bf    = (bf16*)(ws);  ws += nQ * 2;
  bf16* kv_bf   = (bf16*)(ws);  ws += nKV * 2;
  bf16* wqT_bf  = (bf16*)(ws);  ws += nWq * 2;    // (N=1024, K=1024)
  bf16* wkvT_bf = (bf16*)(ws);  ws += nWkv * 2;   // (N=2048, K=1024)
  bf16* woutT_bf= (bf16*)(ws);  ws += nWout * 2;  // (N=1024, K=1024)
  bf16* qp_bf   = (bf16*)(ws);  ws += nQP * 2;
  bf16* kvp_bf  = (bf16*)(ws);  ws += nKVP * 2;
  bf16* att_bf  = (bf16*)(ws);  ws += nATT * 2;

  auto cvt = [&](const float* src, bf16* dst, size_t n) {
    int blocks = (int)((n + 255) / 256);
    if (blocks > 4096) blocks = 4096;
    f32_to_bf16_kernel<<<blocks, 256, 0, stream>>>(src, dst, (int)n);
  };
  cvt(q, q_bf, nQ);
  cvt(kv, kv_bf, nKV);
  // Weights converted + transposed once: (K,N) fp32 -> (N,K) bf16.
  f32_to_bf16_t_kernel<<<4096, 256, 0, stream>>>(w_q, wqT_bf, 1024, 1024);
  f32_to_bf16_t_kernel<<<4096, 256, 0, stream>>>(w_kv, wkvT_bf, 1024, 2048);
  f32_to_bf16_t_kernel<<<4096, 256, 0, stream>>>(w_out, woutT_bf, 1024, 1024);

  // Q projection: (4096,1024) @ (1024,1024) -> bf16 (4096,1024)
  {
    dim3 grid(1024 / 64, 4096 / 128);
    gemm_bf16_tdm_kernel<bf16><<<grid, 256, 0, stream>>>(
        q_bf, wqT_bf, b_q, qp_bf, 4096, 1024, 1024);
  }
  // KV projection: (4096,1024) @ (1024,2048) -> bf16 (4096,2048)
  {
    dim3 grid(2048 / 64, 4096 / 128);
    gemm_bf16_tdm_kernel<bf16><<<grid, 256, 0, stream>>>(
        kv_bf, wkvT_bf, b_kv, kvp_bf, 4096, 2048, 1024);
  }
  // Flash attention: grid (Nq/64, H, B), 128 threads.
  {
    dim3 grid(2048 / 64, 16, 2);
    flash_attn_kernel<<<grid, 128, 0, stream>>>(qp_bf, kvp_bf, att_bf);
  }
  // Output projection: (4096,1024) @ (1024,1024) -> fp32 out
  {
    dim3 grid(1024 / 64, 4096 / 128);
    gemm_bf16_tdm_kernel<float><<<grid, 256, 0, stream>>>(
        att_bf, woutT_bf, b_out, out, 4096, 1024, 1024);
  }
}
